// MultiHeadAttention_30494267802056
// MI455X (gfx1250) — matrix operand order repaired and checked by reading, not compiled, not run on hardware
//
#include <hip/hip_runtime.h>
#include <hip/hip_bf16.h>

#define BB 128
#define TT 256
#define CC 384
#define HH 6
#define HS 64

typedef __bf16 bf16;
typedef __attribute__((ext_vector_type(16))) __bf16 v16bf;
typedef __attribute__((ext_vector_type(8)))  __bf16 v8bf;
typedef __attribute__((ext_vector_type(8)))  float  v8f;
typedef unsigned int u32;
typedef u32 u32x4 __attribute__((ext_vector_type(4)));
typedef u32 u32x8 __attribute__((ext_vector_type(8)));

// ---------------- WMMA fragment helpers (ISA 7.12.2 layouts, wave32) ----------------

// A-matrix 16x32 bf16: lane L holds row m=L&15; K = 8*(L>>4)+j (j<8) and +16 for j>=8.
__device__ __forceinline__ v16bf load_a_frag(const bf16* __restrict__ base, int ld) {
    int lane = threadIdx.x & 31;
    int m  = lane & 15;
    int kb = (lane >> 4) * 8;
    const bf16* p = base + m * ld + kb;
    union { v16bf v; v8bf h[2]; } u;
    u.h[0] = *(const v8bf*)(p);        // K = kb .. kb+7
    u.h[1] = *(const v8bf*)(p + 16);   // K = kb+16 .. kb+23
    return u.v;
}

// B-matrix 32x16 bf16 loaded from a [N][K] (i.e. B-transposed, K-major) buffer:
// lane L holds column n=L&15 of B; K = 8*(L>>4)+j (j<8) and +16 for j>=8.
__device__ __forceinline__ v16bf load_bt_frag(const bf16* __restrict__ base, int ld) {
    int lane = threadIdx.x & 31;
    int n  = lane & 15;
    int kb = (lane >> 4) * 8;
    const bf16* p = base + n * ld + kb;
    union { v16bf v; v8bf h[2]; } u;
    u.h[0] = *(const v8bf*)(p);
    u.h[1] = *(const v8bf*)(p + 16);
    return u.v;
}

__device__ __forceinline__ v8f wmma_bf16(v16bf a, v16bf b, v8f c) {
    return __builtin_amdgcn_wmma_f32_16x16x32_bf16(false, a, false, b, (short)0, c, false, false);
}

// ---------------- gfx1250 async-to-LDS + TDM helpers (CDNA5 ISA) ----------------

__device__ __forceinline__ void async_ld16(unsigned lds_off, const void* g) {
    asm volatile("global_load_async_to_lds_b128 %0, %1, off"
                 :: "v"(lds_off), "v"((unsigned long long)(uintptr_t)g)
                 : "memory");
}
#define WAIT_ASYNC(n)  asm volatile("s_wait_asynccnt " #n ::: "memory")
#define WAIT_TENSOR(n) asm volatile("s_wait_tensorcnt " #n ::: "memory")
#define WAIT_DS0()     asm volatile("s_wait_dscnt 0" ::: "memory")

// TDM 2-D tile load: tile [tile_d1 rows][tile_d0 elems] of 2-byte data, row stride
// stride0 (elements) in global, packed contiguously into LDS at lds_off.
// D# per ISA 8.3/8.4: group0 = {count|..., lds_addr, gaddr_lo, gaddr_hi|type=2},
// group1 = {data_size, tensor dims, tile dims, strides}. All operands wave-uniform.
__device__ __forceinline__ void tdm_load_2d(unsigned lds_off, const void* gaddr,
                                            u32 tensor_d0, u32 tensor_d1,
                                            u32 tile_d0, u32 tile_d1, u32 stride0) {
    unsigned long long ga = (unsigned long long)(uintptr_t)gaddr;
    u32x4 g0;
    g0[0] = 1u;                                             // count=1 (user descriptor)
    g0[1] = lds_off;                                        // lds_addr
    g0[2] = (u32)ga;                                        // global_addr[31:0]
    g0[3] = (u32)((ga >> 32) & 0x01ffffffu) | (2u << 30);   // global_addr[56:32] | type=2
    u32x8 g1;
    g1[0] = 1u << 16;                                       // data_size=1 -> 2 bytes
    g1[1] = (tensor_d0 & 0xffffu) << 16;                    // tensor_dim0[15:0]
    g1[2] = (tensor_d0 >> 16) | ((tensor_d1 & 0xffffu) << 16);
    g1[3] = (tensor_d1 >> 16) | (tile_d0 << 16);            // tile_dim0
    g1[4] = tile_d1;                                        // tile_dim1 (tile_dim2=0)
    g1[5] = stride0;                                        // tensor_dim0_stride[31:0]
    g1[6] = 0;
    g1[7] = 0;
    asm volatile("tensor_load_to_lds %0, %1" :: "s"(g0), "s"(g1) : "memory");
}

// ---------------- prep kernels ----------------

__global__ void k_cast_x(const float* __restrict__ x, bf16* __restrict__ xb, int n) {
    int i = blockIdx.x * blockDim.x + threadIdx.x;
    if (i < n) xb[i] = (bf16)x[i];
}

// W: [H][C][HS] -> Wt: [H][HS][C]  (bf16)
__global__ void k_transpose_w(const float* __restrict__ w, bf16* __restrict__ wt) {
    int i = blockIdx.x * blockDim.x + threadIdx.x;
    if (i >= HH * CC * HS) return;
    int d = i % HS;
    int c = (i / HS) % CC;
    int h = i / (HS * CC);
    wt[(h * HS + d) * CC + c] = (bf16)w[i];
}

// Wp: [J=384][C=384] -> Wpt: [C][J]  (bf16)
__global__ void k_transpose_wp(const float* __restrict__ w, bf16* __restrict__ wt) {
    int i = blockIdx.x * blockDim.x + threadIdx.x;
    if (i >= CC * CC) return;
    int c = i % CC;
    int j = i / CC;
    wt[c * CC + j] = (bf16)w[i];
}

// ---------------- QKV projection: q,k row-major [b,h,t,d]; v transposed [b,h,d,s] ----------------

__global__ void k_qkv(const bf16* __restrict__ xb,
                      const bf16* __restrict__ wqt, const bf16* __restrict__ wkt,
                      const bf16* __restrict__ wvt,
                      bf16* __restrict__ qb, bf16* __restrict__ kb, bf16* __restrict__ vtb) {
    int rt  = blockIdx.x;            // 0..2047 : 16-row tile of the B*T dimension
    int h   = blockIdx.y;            // 0..5
    int sel = blockIdx.z;            // 0=q 1=k 2=v
    int lane = threadIdx.x & 31;
    int bt0 = rt * 16;
    int b = bt0 / TT, t0 = bt0 % TT;

    const bf16* wt  = (sel == 0) ? wqt : (sel == 1) ? wkt : wvt;
    const bf16* wth = wt + h * HS * CC;
    const bf16* arow = xb + bt0 * CC;

    v8f acc[4] = {};
    v16bf a_cur = load_a_frag(arow, CC);
    v16bf b_cur[4];
#pragma unroll
    for (int nt = 0; nt < 4; ++nt) b_cur[nt] = load_bt_frag(wth + (nt * 16) * CC, CC);

#pragma unroll
    for (int kk = 0; kk < CC; kk += 32) {
        v16bf a_nxt = {};
        v16bf b_nxt[4] = {};
        if (kk + 32 < CC) {
            a_nxt = load_a_frag(arow + kk + 32, CC);
#pragma unroll
            for (int nt = 0; nt < 4; ++nt)
                b_nxt[nt] = load_bt_frag(wth + (nt * 16) * CC + kk + 32, CC);
        }
#pragma unroll
        for (int nt = 0; nt < 4; ++nt)
            acc[nt] = wmma_bf16(a_cur, b_cur[nt], acc[nt]);
        a_cur = a_nxt;
#pragma unroll
        for (int nt = 0; nt < 4; ++nt) b_cur[nt] = b_nxt[nt];
    }

    int n = lane & 15, hh = lane >> 4;
    if (sel < 2) {
        bf16* dst = ((sel == 0) ? qb : kb) + ((b * HH + h) * TT + t0) * HS;
#pragma unroll
        for (int nt = 0; nt < 4; ++nt)
#pragma unroll
            for (int r = 0; r < 8; ++r)
                dst[(hh * 8 + r) * HS + nt * 16 + n] = (bf16)acc[nt][r];
    } else {
        // v transposed: vtb[b][h][d][s]; lane's 8 C-elements are 8 consecutive s for fixed d=n
        bf16* dst = vtb + ((b * HH + h) * HS) * TT;
#pragma unroll
        for (int nt = 0; nt < 4; ++nt) {
            v8bf pk;
#pragma unroll
            for (int r = 0; r < 8; ++r) pk[r] = (bf16)acc[nt][r];
            *(v8bf*)(dst + (nt * 16 + n) * TT + t0 + hh * 8) = pk;
        }
    }
}

// ------- fused causal attention: K via async-to-LDS, V via TDM, double-buffered -------

__global__ void k_attn(const bf16* __restrict__ qb, const bf16* __restrict__ kb,
                       const bf16* __restrict__ vtb, bf16* __restrict__ oh) {
    __shared__ __align__(16) bf16 kbuf[2][32 * HS];   // [s_local][d], 4KB each
    __shared__ __align__(16) bf16 vbuf[2][HS * 32];   // [d][s_local], 4KB each
    __shared__ __align__(16) bf16 pl[16 * 32];        // P tile, row-major

    int it = blockIdx.x;     // q tile 0..15
    int h  = blockIdx.y;
    int b  = blockIdx.z;
    int lane = threadIdx.x & 31;
    int n = lane & 15, hh = lane >> 4;

    const bf16* qbh = qb  + ((b * HH + h) * TT) * HS;  // [t][d]
    const bf16* kbh = kb  + ((b * HH + h) * TT) * HS;  // [s][d]
    const bf16* vth = vtb + ((b * HH + h) * HS) * TT;  // [d][s]

    int t0 = it * 16;
    v16bf qa0 = load_a_frag(qbh + t0 * HS + 0,  HS);
    v16bf qa1 = load_a_frag(qbh + t0 * HS + 32, HS);

    float m_i[8], l_i[8];
#pragma unroll
    for (int r = 0; r < 8; ++r) { m_i[r] = -__builtin_inff(); l_i[r] = 0.f; }
    v8f o[4] = {};

    const int nkeys = t0 + 16;        // causal: keys 0 .. t0+15
    const int nblk  = (nkeys + 31) / 32;
    const float scale = 0.125f;       // 1/sqrt(64)

    // stage one K/V block: K = 8 async b128 (4KB contiguous); V = one TDM 2-D tile
    auto stage = [&](int s0, int buf) {
        unsigned klo = (unsigned)(uintptr_t)&kbuf[buf][0];
        const char* kg = (const char*)(kbh + s0 * HS);
#pragma unroll
        for (int i = 0; i < 8; ++i)
            async_ld16(klo + (i * 32 + lane) * 16, kg + (i * 32 + lane) * 16);
        // V tile: 64 rows (d) x 32 elems (s), global row stride TT, packed 64B rows in LDS
        tdm_load_2d((unsigned)(uintptr_t)&vbuf[buf][0], vth + s0,
                    /*tensor_d0=*/TT, /*tensor_d1=*/HS,
                    /*tile_d0=*/32, /*tile_d1=*/HS, /*stride0=*/TT);
    };

    stage(0, 0);

    for (int jb = 0; jb < nblk; ++jb) {
        int s0  = jb * 32;
        int cur = jb & 1;
        if (jb + 1 < nblk) {
            WAIT_DS0();                 // ds reads of the buffer being recycled are done
            stage(s0 + 32, cur ^ 1);
            WAIT_ASYNC(8);              // current K tile landed (8 of 16 outstanding)
            WAIT_TENSOR(1);             // current V tile landed (1 of 2 outstanding)
        } else {
            WAIT_ASYNC(0);
            WAIT_TENSOR(0);
        }
        const bf16* kl = &kbuf[cur][0];
        const bf16* vl = &vbuf[cur][0];

        float p[2][8];
        float rowmax[8];
#pragma unroll
        for (int r = 0; r < 8; ++r) rowmax[r] = -__builtin_inff();

#pragma unroll
        for (int ns = 0; ns < 2; ++ns) {
            int sc = s0 + ns * 16;
            v8f s = {};
            if (sc < nkeys) {
                v16bf kf0 = load_bt_frag(kl + (ns * 16) * HS + 0,  HS);
                v16bf kf1 = load_bt_frag(kl + (ns * 16) * HS + 32, HS);
                s = wmma_bf16(qa0, kf0, s);
                s = wmma_bf16(qa1, kf1, s);
            }
#pragma unroll
            for (int r = 0; r < 8; ++r) {
                int qrow = t0 + hh * 8 + r;
                int col  = sc + n;
                float val = (sc < nkeys && col <= qrow) ? s[r] * scale : -__builtin_inff();
                p[ns][r] = val;
                rowmax[r] = fmaxf(rowmax[r], val);
            }
        }

        // online softmax; row values live across the 16 lanes of each half-wave
#pragma unroll
        for (int r = 0; r < 8; ++r) {
            float v = rowmax[r];
            v = fmaxf(v, __shfl_xor(v, 1, 32));
            v = fmaxf(v, __shfl_xor(v, 2, 32));
            v = fmaxf(v, __shfl_xor(v, 4, 32));
            v = fmaxf(v, __shfl_xor(v, 8, 32));
            float mnew  = fmaxf(m_i[r], v);
            float alpha = __expf(m_i[r] - mnew);
            float p0 = __expf(p[0][r] - mnew);
            float p1 = __expf(p[1][r] - mnew);
            p[0][r] = p0; p[1][r] = p1;
            float rs = p0 + p1;
            rs += __shfl_xor(rs, 1, 32);
            rs += __shfl_xor(rs, 2, 32);
            rs += __shfl_xor(rs, 4, 32);
            rs += __shfl_xor(rs, 8, 32);
            l_i[r] = l_i[r] * alpha + rs;
            m_i[r] = mnew;
#pragma unroll
            for (int nt = 0; nt < 4; ++nt) o[nt][r] *= alpha;
        }

        // C-layout -> LDS row-major -> A-fragment
#pragma unroll
        for (int ns = 0; ns < 2; ++ns)
#pragma unroll
            for (int r = 0; r < 8; ++r)
                pl[(hh * 8 + r) * 32 + ns * 16 + n] = (bf16)p[ns][r];
        __syncthreads();

        v16bf pf = load_a_frag(pl, 32);
#pragma unroll
        for (int nt = 0; nt < 4; ++nt) {
            v16bf vf = load_bt_frag(vl + (nt * 16) * 32, 32);
            o[nt] = wmma_bf16(pf, vf, o[nt]);
        }
        __syncthreads();
    }

    // normalize and write out_heads [b][t][h*64+d]
    bf16* dst = oh + (b * TT + t0) * (HH * HS) + h * HS;
#pragma unroll
    for (int nt = 0; nt < 4; ++nt)
#pragma unroll
        for (int r = 0; r < 8; ++r)
            dst[(hh * 8 + r) * (HH * HS) + nt * 16 + n] = (bf16)(o[nt][r] / l_i[r]);
}

// ---------------- output projection: out = oh @ Wp + bp (f32 out) ----------------

__global__ void k_proj(const bf16* __restrict__ oh, const bf16* __restrict__ wpt,
                       const float* __restrict__ bp, float* __restrict__ out) {
    int rt = blockIdx.x;   // 0..2047 row tile
    int ng = blockIdx.y;   // 0..5  -> 64 output cols
    int lane = threadIdx.x & 31;
    int n = lane & 15, hh = lane >> 4;
    int r0 = rt * 16;
    const bf16* arow = oh + r0 * CC;
    const bf16* wrow = wpt + (ng * 64) * CC;

    v8f acc[4] = {};
    v16bf a_cur = load_a_frag(arow, CC);
    v16bf b_cur[4];
#pragma unroll
    for (int nt = 0; nt < 4; ++nt) b_cur[nt] = load_bt_frag(wrow + (nt * 16) * CC, CC);

#pragma unroll
    for (int kk = 0; kk < CC; kk += 32) {
        v16bf a_nxt = {};
        v16bf b_nxt[4] = {};
        if (kk + 32 < CC) {
            a_nxt = load_a_frag(arow + kk + 32, CC);
#pragma unroll
            for (int nt = 0; nt < 4; ++nt)
                b_nxt[nt] = load_bt_frag(wrow + (nt * 16) * CC + kk + 32, CC);
        }
#pragma unroll
        for (int nt = 0; nt < 4; ++nt)
            acc[nt] = wmma_bf16(a_cur, b_cur[nt], acc[nt]);
        a_cur = a_nxt;
#pragma unroll
        for (int nt = 0; nt < 4; ++nt) b_cur[nt] = b_nxt[nt];
    }

#pragma unroll
    for (int nt = 0; nt < 4; ++nt) {
        int c = ng * 64 + nt * 16 + n;
        float bias = bp[c];
#pragma unroll
        for (int r = 0; r < 8; ++r)
            out[(size_t)(r0 + hh * 8 + r) * CC + c] = acc[nt][r] + bias;
    }
}

// ---------------- host side ----------------

extern "C" void kernel_launch(void* const* d_in, const int* in_sizes, int n_in,
                              void* d_out, int out_size, void* d_ws, size_t ws_size,
                              hipStream_t stream) {
    const float* x  = (const float*)d_in[0];
    const float* Wq = (const float*)d_in[1];
    const float* Wk = (const float*)d_in[2];
    const float* Wv = (const float*)d_in[3];
    const float* Wp = (const float*)d_in[4];
    const float* bp = (const float*)d_in[5];
    float* out = (float*)d_out;

    char* ws = (char*)d_ws;
    size_t off = 0;
    auto alloc = [&](size_t bytes) -> void* {
        void* p = ws + off;
        off += (bytes + 255) & ~(size_t)255;
        return p;
    };

    bf16* xb  = (bf16*)alloc((size_t)BB * TT * CC * sizeof(bf16));
    bf16* qb  = (bf16*)alloc((size_t)BB * HH * TT * HS * sizeof(bf16));
    bf16* kb  = (bf16*)alloc((size_t)BB * HH * TT * HS * sizeof(bf16));
    bf16* vtb = (bf16*)alloc((size_t)BB * HH * HS * TT * sizeof(bf16));
    bf16* oh  = (bf16*)alloc((size_t)BB * TT * HH * HS * sizeof(bf16));
    bf16* wqt = (bf16*)alloc((size_t)HH * HS * CC * sizeof(bf16));
    bf16* wkt = (bf16*)alloc((size_t)HH * HS * CC * sizeof(bf16));
    bf16* wvt = (bf16*)alloc((size_t)HH * HS * CC * sizeof(bf16));
    bf16* wpt = (bf16*)alloc((size_t)CC * CC * sizeof(bf16));

    int n = BB * TT * CC;
    k_cast_x<<<(n + 255) / 256, 256, 0, stream>>>(x, xb, n);
    n = HH * CC * HS;
    k_transpose_w<<<(n + 255) / 256, 256, 0, stream>>>(Wq, wqt);
    k_transpose_w<<<(n + 255) / 256, 256, 0, stream>>>(Wk, wkt);
    k_transpose_w<<<(n + 255) / 256, 256, 0, stream>>>(Wv, wvt);
    n = CC * CC;
    k_transpose_wp<<<(n + 255) / 256, 256, 0, stream>>>(Wp, wpt);

    dim3 gq(BB * TT / 16, HH, 3);
    k_qkv<<<gq, 32, 0, stream>>>(xb, wqt, wkt, wvt, qb, kb, vtb);

    dim3 ga(TT / 16, HH, BB);
    k_attn<<<ga, 32, 0, stream>>>(qb, kb, vtb, oh);

    dim3 gp(BB * TT / 16, HH);
    k_proj<<<gp, 32, 0, stream>>>(oh, wpt, bp, out);
}
